// PrototypeConditionedStageBlock_24446953849146
// MI455X (gfx1250) — matrix operand order repaired and checked by reading, not compiled, not run on hardware
//
#include <hip/hip_runtime.h>
#include <hip/hip_bf16.h>
#include <math.h>

// ---------------------------------------------------------------------------
// PrototypeConditionedStageBlock for MI455X (gfx1250, wave32, WMMA + TDM).
//
// Compute-bound (143 GFLOP vs ~125MB unique traffic => ~5.4us at 23.3TB/s HBM):
// all heavy GEMMs go through v_wmma_f32_16x16x32_bf16. Weights are converted
// fp32->bf16 and transposed to N-major once (36MB bf16 total, fits entirely in
// the 192MB L2). Operand tiles (128x32 A and B) are DMA'd into LDS by the
// Tensor Data Mover (tensor_load_to_lds, TENSORcnt, double-buffered), and the
// waves feed WMMA from LDS via ds_load_b128 using the ISA fragment layouts.
// ---------------------------------------------------------------------------

typedef __attribute__((ext_vector_type(16))) __bf16 v16bf;
typedef __attribute__((ext_vector_type(8)))  float  v8f;
typedef unsigned int u32x4 __attribute__((ext_vector_type(4)));
typedef int          i32x4 __attribute__((ext_vector_type(4)));
typedef int          i32x8 __attribute__((ext_vector_type(8)));

namespace {
constexpr int BB   = 2;
constexpr int SS   = 2048;
constexpr int NTOK = BB * SS;     // 4096
constexpr int Dk   = 1024;
constexpr int NFk  = 32;
constexpr int Pk   = 256;
constexpr int FEk  = 128;
constexpr int RIN  = Dk + FEk;    // 1152
constexpr int RHk  = 256;
constexpr int Ek   = 8;
constexpr int EHk  = 1024;

constexpr size_t alignup(size_t x) { return (x + 255) & ~(size_t)255; }

// workspace layout (bytes)
constexpr size_t OFF_CTXH = 0;                                              // [2][1024] f32
constexpr size_t OFF_CTXF = alignup(OFF_CTXH + (size_t)BB * Dk * 4);        // [2][32]   f32
constexpr size_t OFF_XB   = alignup(OFF_CTXF + (size_t)BB * NFk * 4);       // [4096][1152] bf16
constexpr size_t OFF_WR1T = alignup(OFF_XB   + (size_t)NTOK * RIN * 2);     // [256][1152]  bf16
constexpr size_t OFF_XR   = alignup(OFF_WR1T + (size_t)RHk * RIN * 2);      // [4096][256]  bf16
constexpr size_t OFF_WE1T = alignup(OFF_XR   + (size_t)NTOK * RHk * 2);     // [8][1024][1152] bf16
constexpr size_t OFF_WE2T = alignup(OFF_WE1T + (size_t)Ek * EHk * RIN * 2); // [8][1024][1024] bf16
constexpr size_t OFF_HBUF = alignup(OFF_WE2T + (size_t)Ek * Dk * EHk * 2);  // [4096][1024] bf16 (reused per expert)

// d_out element offsets: stage_delta | gate_weights | gate_logits
constexpr size_t OUT_GW = (size_t)NTOK * Dk;
constexpr size_t OUT_GL = OUT_GW + (size_t)NTOK * Ek;

__device__ __forceinline__ float gelu_f(float x) {
  // tanh approximation (jax.nn.gelu default)
  const float c0 = 0.7978845608028654f, c1 = 0.044715f;
  return 0.5f * x * (1.0f + tanhf(c0 * (x + c1 * x * x * x)));
}

// ---------------------------------------------------------------------------
// TDM: DMA a [128 rows x 32 elements] bf16 tile (row stride = lineLen elems)
// from global memory into contiguous LDS at lds_addr. 2-D descriptor:
// groups 2/3 are zero (VADDR2/3-style unused dims), workgroup_mask=0 (no
// cluster multicast). Tracked by TENSORcnt.
// ---------------------------------------------------------------------------
__device__ __forceinline__ void tdm_load_tile_128x32(
    unsigned lds_addr, const __hip_bfloat16* gptr, int lineLen, int nRows) {
  const unsigned long long ga = (unsigned long long)(uintptr_t)gptr;
  u32x4 g0 = {0u, 0u, 0u, 0u};
  g0[0] = 1u;                                   // count=1, user descriptor
  g0[1] = lds_addr;                             // lds_addr at bits [63:32]
  g0[2] = (unsigned)(ga & 0xffffffffull);       // global_addr[31:0]
  g0[3] = (unsigned)((ga >> 32) & 0x1ffffffull) // global_addr[56:32]
          | (2u << 30);                         // type=2 ("image")
  const unsigned td0 = (unsigned)lineLen;       // tensor_dim0 (elements)
  const unsigned td1 = (unsigned)nRows;         // tensor_dim1 (rows)
  const unsigned long long st0 = (unsigned long long)lineLen; // dim0 stride
  i32x8 g1 = {0, 0, 0, 0, 0, 0, 0, 0};
  g1[0] = (int)(1u << 16);                      // data_size=1 (2 bytes)
  g1[1] = (int)((td0 & 0xffffu) << 16);         // tensor_dim0[15:0] @ [63:48]
  g1[2] = (int)((td0 >> 16) | ((td1 & 0xffffu) << 16)); // td0 hi | td1 lo
  g1[3] = (int)((td1 >> 16) | (32u << 16));     // td1 hi | tile_dim0=32
  g1[4] = (int)(128u);                          // tile_dim1=128, tile_dim2=0
  g1[5] = (int)(st0 & 0xffffffffull);           // tensor_dim0_stride[31:0]
  g1[6] = (int)((st0 >> 32) & 0xffffull);       // stride[47:32] | dim1_stride=0
  g1[7] = 0;
  i32x4 gz = {0, 0, 0, 0};
#if __clang_major__ >= 23
  i32x8 gz8 = {0, 0, 0, 0, 0, 0, 0, 0};
  __builtin_amdgcn_tensor_load_to_lds(g0, g1, gz, gz, gz8, 0);
#else
  __builtin_amdgcn_tensor_load_to_lds(g0, g1, gz, gz, 0);
#endif
}
} // namespace

// ---------------------------------------------------------------------------
// 1) prototype context projections: ctx_h[b] = proto[b] @ W_hid_ctx,
//    ctx_f[b] = proto[b] @ W_feat_ctx
// ---------------------------------------------------------------------------
__global__ __launch_bounds__(256) void k_ctx(
    const float* __restrict__ proto, const float* __restrict__ Whc,
    const float* __restrict__ Wfc, float* __restrict__ ctx_h,
    float* __restrict__ ctx_f) {
  const int b = blockIdx.x;
  const float* pr = proto + (size_t)b * Pk;
  for (int o = threadIdx.x; o < Dk + NFk; o += blockDim.x) {
    float s = 0.f;
    if (o < Dk) {
      for (int p = 0; p < Pk; ++p) s += pr[p] * Whc[(size_t)p * Dk + o];
      ctx_h[(size_t)b * Dk + o] = s;
    } else {
      const int c = o - Dk;
      for (int p = 0; p < Pk; ++p) s += pr[p] * Wfc[(size_t)p * NFk + c];
      ctx_f[(size_t)b * NFk + c] = s;
    }
  }
}

// ---------------------------------------------------------------------------
// 2) fp32 [batch][R][C] -> bf16 transposed [batch][C][R]
// ---------------------------------------------------------------------------
__global__ __launch_bounds__(256) void k_cvt_transpose(
    const float* __restrict__ in, __hip_bfloat16* __restrict__ out,
    int R, int C, int batch) {
  const size_t total = (size_t)batch * R * C;
  for (size_t idx = (size_t)blockIdx.x * blockDim.x + threadIdx.x; idx < total;
       idx += (size_t)gridDim.x * blockDim.x) {
    const size_t mat = idx / ((size_t)R * C);
    const size_t rem = idx - mat * (size_t)R * C;
    const int r = (int)(rem / C);
    const int c = (int)(rem - (size_t)r * C);
    out[mat * (size_t)R * C + (size_t)c * R + r] = __float2bfloat16(in[idx]);
  }
}

// ---------------------------------------------------------------------------
// 3) build x = [hidden + ctx_h | gelu((feat + ctx_f) @ W_feat_emb + b)] as bf16
// ---------------------------------------------------------------------------
__global__ __launch_bounds__(256) void k_buildx(
    const float* __restrict__ hidden, const float* __restrict__ feat,
    const float* __restrict__ ctx_h, const float* __restrict__ ctx_f,
    const float* __restrict__ Wfe, const float* __restrict__ bfe,
    __hip_bfloat16* __restrict__ Xb) {
  const int t = blockIdx.x;
  const int b = t / SS;
  __shared__ float fc[NFk];
  if (threadIdx.x < NFk)
    fc[threadIdx.x] = feat[(size_t)t * NFk + threadIdx.x] + ctx_f[(size_t)b * NFk + threadIdx.x];
  __syncthreads();

  const float* hr = hidden + (size_t)t * Dk;
  const float* ch = ctx_h + (size_t)b * Dk;
  __hip_bfloat16* xr = Xb + (size_t)t * RIN;
  for (int i = threadIdx.x; i < Dk; i += 256)
    xr[i] = __float2bfloat16(hr[i] + ch[i]);
  if (threadIdx.x < FEk) {
    float s = bfe[threadIdx.x];
    #pragma unroll
    for (int j = 0; j < NFk; ++j) s += fc[j] * Wfe[(size_t)j * FEk + threadIdx.x];
    xr[Dk + threadIdx.x] = __float2bfloat16(gelu_f(s));
  }
}

// ---------------------------------------------------------------------------
// WMMA bf16 GEMM: C[M,N] = A[M,K] @ BT[N,K]^T (+bias), two epilogues:
//   mode 0: outB = bf16(gelu(C + bias))
//   mode 1: outF (+)= gate[row*8] * (C + bias[col])   (gate-weighted combine)
// Block = 256 threads = 8 waves (4 M x 2 N); wave tile 32x64 = 2x4 WMMA frags.
// Block tile 128x128, K step 32. A/B tiles are TDM-DMA'd into double-buffered
// LDS by wave 0 (TDM ignores EXEC -> wave-uniform guard avoids 8x issue),
// published with s_wait_tensorcnt + workgroup barrier.
// ---------------------------------------------------------------------------
union Frag {
  v16bf v;
  uint4 q[2];
};

__global__ __launch_bounds__(256) void gemm_wmma_bf16(
    const __hip_bfloat16* __restrict__ A, const __hip_bfloat16* __restrict__ BT,
    const float* __restrict__ bias, __hip_bfloat16* __restrict__ outB,
    float* __restrict__ outF, const float* __restrict__ gate,
    int M, int N, int K, int mode, int initOut) {
  __shared__ __align__(16) __hip_bfloat16 sA[2][128 * 32];
  __shared__ __align__(16) __hip_bfloat16 sB[2][128 * 32];

  const int lane  = threadIdx.x & 31;
  const int wave  = threadIdx.x >> 5;
  const int waveM = wave >> 1;  // 0..3
  const int waveN = wave & 1;   // 0..1
  const int hlf   = lane >> 4;  // lane half selects K sub-pattern (ISA layout)
  const int l15   = lane & 15;

  const int blockM = blockIdx.y * 128;
  const int blockN = blockIdx.x * 128;
  const int m0 = blockM + waveM * 32;
  const int n0 = blockN + waveN * 64;

  const unsigned ldsA[2] = {(unsigned)(uintptr_t)&sA[0][0],
                            (unsigned)(uintptr_t)&sA[1][0]};
  const unsigned ldsB[2] = {(unsigned)(uintptr_t)&sB[0][0],
                            (unsigned)(uintptr_t)&sB[1][0]};
  const __hip_bfloat16* gA = A + (size_t)blockM * K;   // tile rows start
  const __hip_bfloat16* gB = BT + (size_t)blockN * K;

  v8f acc[2][4] = {};

  // prologue: TDM-issue K-chunk 0 into buffer 0
  if (wave == 0) {
    tdm_load_tile_128x32(ldsA[0], gA, K, M);
    tdm_load_tile_128x32(ldsB[0], gB, K, N);
  }

  int buf = 0;
  for (int kk = 0; kk < K; kk += 32) {
    if (wave == 0) {
      if (kk + 32 < K) {  // double-buffer: launch next pair, wait current pair
        tdm_load_tile_128x32(ldsA[buf ^ 1], gA + (kk + 32), K, M);
        tdm_load_tile_128x32(ldsB[buf ^ 1], gB + (kk + 32), K, N);
        __builtin_amdgcn_s_wait_tensorcnt(2);
      } else {
        __builtin_amdgcn_s_wait_tensorcnt(0);
      }
    }
    __syncthreads();  // publish TDM-filled LDS tile to all 8 waves

    // A fragment: row-local = waveM*32 + i*16 + l15; K {0..7,16..23} + 8*half
    Frag a[2];
    #pragma unroll
    for (int i = 0; i < 2; ++i) {
      const uint4* p = (const uint4*)(&sA[buf][(waveM * 32 + i * 16 + l15) * 32 + 8 * hlf]);
      a[i].q[0] = p[0];  // 8 bf16 @ K+0
      a[i].q[1] = p[2];  // 8 bf16 @ K+16
    }
    // B fragment: row-local = waveN*64 + j*16 + l15; K {0..7,16..23} + 8*half
    Frag b[4];
    #pragma unroll
    for (int j = 0; j < 4; ++j) {
      const uint4* p = (const uint4*)(&sB[buf][(waveN * 64 + j * 16 + l15) * 32 + 8 * hlf]);
      b[j].q[0] = p[0];
      b[j].q[1] = p[2];
    }
    #pragma unroll
    for (int i = 0; i < 2; ++i)
      #pragma unroll
      for (int j = 0; j < 4; ++j)
        acc[i][j] = __builtin_amdgcn_wmma_f32_16x16x32_bf16(
            false, a[i].v, false, b[j].v, (short)0, acc[i][j], false, false);

    __syncthreads();  // all reads done before this buffer is re-filled
    buf ^= 1;
  }

  // epilogue: C/D layout -> M = vgpr + 8*half, N = lane&15
  #pragma unroll
  for (int i = 0; i < 2; ++i) {
    #pragma unroll
    for (int j = 0; j < 4; ++j) {
      const int colg = n0 + j * 16 + l15;
      const float bv = bias[colg];
      #pragma unroll
      for (int r = 0; r < 8; ++r) {
        const int rowg = m0 + i * 16 + 8 * hlf + r;
        const float v = acc[i][j][r] + bv;
        if (mode == 0) {
          outB[(size_t)rowg * N + colg] = __float2bfloat16(gelu_f(v));
        } else {
          const float g = gate[(size_t)rowg * 8];
          float* dst = outF + (size_t)rowg * N + colg;
          const float res = g * v;
          if (initOut) *dst = res; else *dst += res;
        }
      }
    }
  }
}

// ---------------------------------------------------------------------------
// 5) router layer 2 + softmax + top-2 renormalize. 8 threads per token.
// ---------------------------------------------------------------------------
__global__ __launch_bounds__(256) void k_router2(
    const __hip_bfloat16* __restrict__ XR, const float* __restrict__ Wr2,
    const float* __restrict__ br2, float* __restrict__ gw_out,
    float* __restrict__ gl_out) {
  __shared__ float lg[32][Ek];
  const int tl = threadIdx.x >> 3;
  const int e  = threadIdx.x & 7;
  const int tok = blockIdx.x * 32 + tl;
  const __hip_bfloat16* x = XR + (size_t)tok * RHk;
  float s = br2[e];
  for (int k = 0; k < RHk; ++k)
    s += __bfloat162float(x[k]) * Wr2[(size_t)k * Ek + e];
  // TEMP == 1.0
  lg[tl][e] = s;
  gl_out[(size_t)tok * Ek + e] = s;
  __syncthreads();
  if (e == 0) {
    float m = lg[tl][0];
    #pragma unroll
    for (int j = 1; j < Ek; ++j) m = fmaxf(m, lg[tl][j]);
    float g[Ek];
    float sum = 0.f;
    #pragma unroll
    for (int j = 0; j < Ek; ++j) { g[j] = expf(lg[tl][j] - m); sum += g[j]; }
    const float inv = 1.f / sum;
    #pragma unroll
    for (int j = 0; j < Ek; ++j) g[j] *= inv;
    int i0 = 0;
    for (int j = 1; j < Ek; ++j) if (g[j] > g[i0]) i0 = j;           // first max
    int i1 = (i0 == 0) ? 1 : 0;
    for (int j = 0; j < Ek; ++j) if (j != i0 && g[j] > g[i1]) i1 = j; // second
    const float denom = g[i0] + g[i1];
    for (int j = 0; j < Ek; ++j) {
      float w = 0.f;
      if (j == i0) w = g[i0] / denom;
      else if (j == i1) w = g[i1] / denom;
      gw_out[(size_t)tok * Ek + j] = w;
    }
  }
}

// ---------------------------------------------------------------------------
extern "C" void kernel_launch(void* const* d_in, const int* in_sizes, int n_in,
                              void* d_out, int out_size, void* d_ws, size_t ws_size,
                              hipStream_t stream) {
  (void)in_sizes; (void)n_in; (void)out_size; (void)ws_size;
  const float* hidden   = (const float*)d_in[0];
  const float* feat     = (const float*)d_in[1];
  const float* proto    = (const float*)d_in[2];
  const float* Whc      = (const float*)d_in[3];
  const float* Wfc      = (const float*)d_in[4];
  const float* Wfe      = (const float*)d_in[5];
  const float* bfe      = (const float*)d_in[6];
  const float* Wr1      = (const float*)d_in[7];
  const float* br1      = (const float*)d_in[8];
  const float* Wr2      = (const float*)d_in[9];
  const float* br2      = (const float*)d_in[10];
  const float* We1      = (const float*)d_in[11];
  const float* be1      = (const float*)d_in[12];
  const float* We2      = (const float*)d_in[13];
  const float* be2      = (const float*)d_in[14];
  float* out = (float*)d_out;

  char* ws = (char*)d_ws;
  float*          ctx_h = (float*)(ws + OFF_CTXH);
  float*          ctx_f = (float*)(ws + OFF_CTXF);
  __hip_bfloat16* Xb    = (__hip_bfloat16*)(ws + OFF_XB);
  __hip_bfloat16* Wr1T  = (__hip_bfloat16*)(ws + OFF_WR1T);
  __hip_bfloat16* XR    = (__hip_bfloat16*)(ws + OFF_XR);
  __hip_bfloat16* We1T  = (__hip_bfloat16*)(ws + OFF_WE1T);
  __hip_bfloat16* We2T  = (__hip_bfloat16*)(ws + OFF_WE2T);
  __hip_bfloat16* Hbuf  = (__hip_bfloat16*)(ws + OFF_HBUF);

  const dim3 blk(256);

  // 1) context projections
  k_ctx<<<dim3(BB), blk, 0, stream>>>(proto, Whc, Wfc, ctx_h, ctx_f);

  // 2) weight convert + transpose (fp32 -> bf16, N-major); all fit in 192MB L2
  k_cvt_transpose<<<dim3(1024), blk, 0, stream>>>(Wr1, Wr1T, RIN, RHk, 1);
  k_cvt_transpose<<<dim3(4096), blk, 0, stream>>>(We1, We1T, RIN, EHk, Ek);
  k_cvt_transpose<<<dim3(4096), blk, 0, stream>>>(We2, We2T, EHk, Dk, Ek);

  // 3) build router/expert input x (bf16 [4096 x 1152])
  k_buildx<<<dim3(NTOK), blk, 0, stream>>>(hidden, feat, ctx_h, ctx_f, Wfe, bfe, Xb);

  // 4) router layer 1: XR = gelu(x @ Wr1 + br1)  [4096 x 256]
  gemm_wmma_bf16<<<dim3(RHk / 128, NTOK / 128), blk, 0, stream>>>(
      Xb, Wr1T, br1, XR, nullptr, nullptr, NTOK, RHk, RIN, 0, 0);

  // 5) router layer 2 + softmax + top-2 -> gate_weights / gate_logits in d_out
  k_router2<<<dim3(NTOK / 32), blk, 0, stream>>>(XR, Wr2, br2, out + OUT_GW, out + OUT_GL);

  // 6) experts (dense, gate-weighted combine). Hbuf reused across experts via
  //    stream ordering; e==0 initializes stage_delta (poison-safe).
  for (int e = 0; e < Ek; ++e) {
    gemm_wmma_bf16<<<dim3(EHk / 128, NTOK / 128), blk, 0, stream>>>(
        Xb, We1T + (size_t)e * EHk * RIN, be1 + (size_t)e * EHk, Hbuf, nullptr,
        nullptr, NTOK, EHk, RIN, 0, 0);
    gemm_wmma_bf16<<<dim3(Dk / 128, NTOK / 128), blk, 0, stream>>>(
        Hbuf, We2T + (size_t)e * Dk * EHk, be2 + (size_t)e * Dk, nullptr, out,
        out + OUT_GW + e, NTOK, Dk, EHk, 1, e == 0 ? 1 : 0);
  }
}
